// VectorQuantizerEMA_55061480735064
// MI455X (gfx1250) — matrix operand order repaired and checked by reading, not compiled, not run on hardware
//
#include <hip/hip_runtime.h>
#include <hip/hip_bf16.h>

// ---------------------------------------------------------------------------
// VQ-VAE EMA vector quantizer for MI455X (gfx1250, wave32, WMMA + async LDS).
//
// argmin_j ||x_n - e_j||^2 == argmin_j (||e_j||^2 - 2 x_n.e_j)
//  -> bf16 WMMA GEMM (v_wmma_f32_16x16x32_bf16), fp32 accumulate,
//     exact fp32 code norms, row norms dropped (argmin-invariant).
// Codebook (512 KB bf16) is staged per-workgroup into LDS with
// global_load_async_to_lds_b128 (double-buffered, ASYNCcnt-paced) so all
// 8 waves share one copy: 8x less L2 traffic than per-wave global reads.
// ---------------------------------------------------------------------------

typedef __attribute__((ext_vector_type(16))) __bf16 v16bf;
typedef __attribute__((ext_vector_type(8)))  float  v8f;

#define N_PTS      65536
#define C_DIM      256
#define K_CODES    1024
#define HW         1024          // H*W
#define Z_ELEMS    16777216      // B*C*H*W
#define DECAY      0.99f
#define ONE_M_D    0.01f
#define EPS_F      1e-5f

#define STAGE_BYTES 32768        // 64 codes * 256 ch * 2B
#define N_STAGES    16           // 16 stages * 64 codes = 1024 codes
#define LDS_BYTES   (2 * STAGE_BYTES)

static __device__ __forceinline__ __bf16 f2bf(float f) {
  union { float f; unsigned u; } a; a.f = f;
  unsigned r = a.u + 0x7FFFu + ((a.u >> 16) & 1u);   // round-to-nearest-even
  union { unsigned short s; __bf16 b; } o;
  o.s = (unsigned short)(r >> 16);
  return o.b;
}

// 16B global -> LDS async copy (ASYNCcnt-tracked). ldsOff is the byte offset
// within the workgroup's LDS allocation; gaddr a 64-bit global address.
static __device__ __forceinline__ void async_cp16(unsigned ldsOff,
                                                  unsigned long long gaddr) {
  asm volatile("global_load_async_to_lds_b128 %0, %1, off"
               :: "v"(ldsOff), "v"(gaddr) : "memory");
}

// ---------------------------------------------------------------------------
// Kernel 1: embedding fp32 -> bf16, code norms ||e_k||^2, zero accumulators.
// ---------------------------------------------------------------------------
__global__ void vq_prep(const float* __restrict__ emb,
                        __bf16* __restrict__ embbf,
                        float*  __restrict__ norms,
                        float*  __restrict__ counts_acc,
                        float*  __restrict__ sums_acc) {
  __shared__ float red[256];
  const int k = blockIdx.x, t = threadIdx.x;
  const size_t idx = ((size_t)k << 8) + t;
  const float e = emb[idx];
  embbf[idx]    = f2bf(e);
  sums_acc[idx] = 0.0f;                      // deterministic accumulators
  red[t] = e * e;
  __syncthreads();
  #pragma unroll
  for (int s = 128; s > 0; s >>= 1) {
    if (t < s) red[t] += red[t + s];
    __syncthreads();
  }
  if (t == 0) { norms[k] = red[0]; counts_acc[k] = 0.0f; }
}

// ---------------------------------------------------------------------------
// Kernel 2: out0 = z (straight-through output equals z in value).
// ---------------------------------------------------------------------------
__global__ void vq_copy(const float4* __restrict__ src, float4* __restrict__ dst,
                        int n4) {
  const int i = blockIdx.x * blockDim.x + threadIdx.x;
  if (i < n4) dst[i] = src[i];
}

// ---------------------------------------------------------------------------
// Kernel 3: per-wave 16 rows x all 1024 codes via bf16 WMMA.
//   blockDim = 256 (8 waves), grid = N/128 = 512. Dynamic LDS = 64 KB.
//   flat[n][c] = z[b*C*HW + c*HW + hw], n = b*HW + hw.
// ---------------------------------------------------------------------------
__global__ void vq_main(const float*  __restrict__ z,
                        const __bf16* __restrict__ embbf,
                        const float*  __restrict__ norms,
                        float* __restrict__ out_idx,
                        float* __restrict__ counts_acc,
                        float* __restrict__ sums_acc) {
  extern __shared__ __align__(32) char smem[];   // 2 x 32 KB codebook stages
  const unsigned ldsBase = (unsigned)(size_t)smem;

  const int lane = threadIdx.x & 31;
  const int wave = threadIdx.x >> 5;
  const int half = lane >> 4;        // 0: lanes 0-15, 1: lanes 16-31
  const int mrow = lane & 15;
  const int rowBase = (blockIdx.x * (blockDim.x >> 5) + wave) << 4;

  // ---- Build 8 A-fragments (16x32 bf16 each) in contiguous k order (k = 16*half + i).
  // A 16-bit 16x32: lane L(0-15) row M=L holds K = 0..15 + kBase,
  //                 lane L+16 same row holds K = 16..31 + kBase.
  const int nA = rowBase + mrow;
  const float* __restrict__ zrow = z + ((size_t)(nA >> 10) << 18) + (nA & (HW - 1));
  v16bf afrag[8];
  #pragma unroll
  for (int ks = 0; ks < 8; ++ks) {
    const int kBase = ks << 5;
    #pragma unroll
    for (int i = 0; i < 16; ++i) {
      const int c = kBase + (half << 4) + i;
      afrag[ks][i] = f2bf(zrow[(size_t)c << 10]);
    }
  }

  float minV[8];
  int   minIdx[8];
  #pragma unroll
  for (int i = 0; i < 8; ++i) { minV[i] = 3.4e38f; minIdx[i] = 0; }

  // ---- Async-stage the bf16 codebook through LDS, double-buffered.
  const unsigned long long gsrc = (unsigned long long)(size_t)embbf;
  const unsigned tOff = (unsigned)threadIdx.x * 16u;

  // prologue: stage 0 -> buffer 0 (8 x b128 per thread = 32 KB per block)
  #pragma unroll
  for (int i = 0; i < 8; ++i)
    async_cp16(ldsBase + tOff + i * 4096u,
               gsrc + tOff + (unsigned)i * 4096u);

  for (int st = 0; st < N_STAGES; ++st) {
    if (st + 1 < N_STAGES) {
      const unsigned bufNxt = ldsBase + (unsigned)((st + 1) & 1) * STAGE_BYTES;
      const unsigned long long gs = gsrc + (unsigned long long)(st + 1) * STAGE_BYTES;
      #pragma unroll
      for (int i = 0; i < 8; ++i)
        async_cp16(bufNxt + tOff + i * 4096u, gs + tOff + (unsigned)i * 4096u);
      // completion is in-order: <=8 outstanding guarantees stage st landed
      asm volatile("s_wait_asynccnt 0x8" ::: "memory");
    } else {
      asm volatile("s_wait_asynccnt 0x0" ::: "memory");
    }
    __syncthreads();

    const __bf16* __restrict__ buf =
        (const __bf16*)(smem + (size_t)(st & 1) * STAGE_BYTES);

    #pragma unroll
    for (int tl = 0; tl < 4; ++tl) {             // 4 column tiles per stage
      const int t  = (st << 2) + tl;
      const int j  = (t << 4) + mrow;            // this lane's global column
      const int jl = (tl << 4) + mrow;           // column within stage
      // B 16-bit 32x16: lane L(0-15) col N=L holds K=0..15 (contiguous),
      //                 lane L+16 same col holds K=16..31.
      const __bf16* bb = buf + ((size_t)jl << 8) + (half << 4);
      v8f acc = {};
      #pragma unroll
      for (int ks = 0; ks < 8; ++ks) {
        const v16bf bfrag = *reinterpret_cast<const v16bf*>(bb + (ks << 5));
        acc = __builtin_amdgcn_wmma_f32_16x16x32_bf16(
            /*neg_a=*/false, afrag[ks], /*neg_b=*/false, bfrag,
            /*c_mod=*/(short)0, acc, /*reuse_a=*/false, /*reuse_b=*/false);
      }
      const float cn = norms[j];
      #pragma unroll
      for (int i = 0; i < 8; ++i) {
        const float s = fmaf(-2.0f, acc[i], cn);         // ||e||^2 - 2 x.e
        if (s < minV[i]) { minV[i] = s; minIdx[i] = j; } // ties -> lowest j
      }
    }
    __syncthreads();
  }

  // ---- Reduce across the 16 lanes sharing each row (stay inside halves).
  #pragma unroll
  for (int i = 0; i < 8; ++i) {
    #pragma unroll
    for (int off = 8; off >= 1; off >>= 1) {
      const float ov = __shfl_xor(minV[i], off, 32);
      const int   oi = __shfl_xor(minIdx[i], off, 32);
      if (ov < minV[i] || (ov == minV[i] && oi < minIdx[i])) {
        minV[i] = ov; minIdx[i] = oi;
      }
    }
  }

  // lanes 0-15 now all hold rows rowBase+0..7; lanes 16-31 rows rowBase+8..15
  if (mrow == 0) {
    #pragma unroll
    for (int i = 0; i < 8; ++i) {
      const int code = minIdx[i];
      out_idx[rowBase + (half << 3) + i] = (float)code;
      atomicAdd(&counts_acc[code], 1.0f);
    }
  }

  // ---- Scatter sums[code] += x_n (fp32 read from z, global fp32 atomics).
  #pragma unroll
  for (int r = 0; r < 16; ++r) {
    const int code = __shfl(minIdx[r & 7], (r < 8) ? 0 : 16, 32);
    const int nr = rowBase + r;
    const float* __restrict__ zr = z + ((size_t)(nr >> 10) << 18) + (nr & (HW - 1));
    float* __restrict__ srow = sums_acc + ((size_t)code << 8);
    #pragma unroll
    for (int ccBase = 0; ccBase < C_DIM; ccBase += 32) {
      const int cc = ccBase + lane;
      atomicAdd(&srow[cc], zr[(size_t)cc << 10]);
    }
  }
}

// ---------------------------------------------------------------------------
// Kernel 4: EMA updates + normalized embedding. One block per code.
// ---------------------------------------------------------------------------
__global__ void vq_final(const float* __restrict__ ema_count,
                         const float* __restrict__ ema_sum,
                         const float* __restrict__ counts_acc,
                         const float* __restrict__ sums_acc,
                         float* __restrict__ out_emb,
                         float* __restrict__ out_count,
                         float* __restrict__ out_sum) {
  const int k = blockIdx.x, t = threadIdx.x;
  const size_t idx = ((size_t)k << 8) + t;
  const float ncnt = ema_count[k] * DECAY + ONE_M_D * counts_acc[k];
  const float nsum = ema_sum[idx] * DECAY + ONE_M_D * sums_acc[idx];
  out_sum[idx] = nsum;
  out_emb[idx] = (nsum + EPS_F) / (ncnt + EPS_F);
  if (t == 0) out_count[k] = ncnt;
}

// ---------------------------------------------------------------------------
extern "C" void kernel_launch(void* const* d_in, const int* in_sizes, int n_in,
                              void* d_out, int out_size, void* d_ws, size_t ws_size,
                              hipStream_t stream) {
  (void)in_sizes; (void)n_in; (void)out_size; (void)ws_size;
  const float* z         = (const float*)d_in[0];   // [64,256,32,32]
  const float* embedding = (const float*)d_in[1];   // [1024,256]
  const float* ema_count = (const float*)d_in[2];   // [1024]
  const float* ema_sum   = (const float*)d_in[3];   // [1024,256]

  float* out = (float*)d_out;
  float* out_z     = out;                                  // 16777216
  float* out_idx   = out + (size_t)Z_ELEMS;                //    65536
  float* out_emb   = out_idx + N_PTS;                      //   262144
  float* out_count = out_emb + (size_t)K_CODES * C_DIM;    //     1024
  float* out_sum   = out_count + K_CODES;                  //   262144

  char* ws = (char*)d_ws;
  __bf16* embbf      = (__bf16*)(ws);                        // 512 KB
  float*  norms      = (float*)(ws + 524288);                //   4 KB
  float*  counts_acc = (float*)(ws + 528384);                //   4 KB
  float*  sums_acc   = (float*)(ws + 532480);                //   1 MB

  // 1) codebook prep: bf16 convert + norms + zero accumulators
  vq_prep<<<K_CODES, 256, 0, stream>>>(embedding, embbf, norms,
                                       counts_acc, sums_acc);
  // 2) straight-through output == z
  vq_copy<<<(Z_ELEMS / 4 + 255) / 256, 256, 0, stream>>>(
      (const float4*)z, (float4*)out_z, Z_ELEMS / 4);
  // 3) WMMA distance argmin + scatter stats; 64 KB dynamic LDS double buffer
  vq_main<<<N_PTS / 128, 256, LDS_BYTES, stream>>>(z, embbf, norms, out_idx,
                                                   counts_acc, sums_acc);
  // 4) EMA finalize
  vq_final<<<K_CODES, 256, 0, stream>>>(ema_count, ema_sum, counts_acc,
                                        sums_acc, out_emb, out_count, out_sum);
}
